// InfiniBlock_42494406426645
// MI455X (gfx1250) — matrix operand order repaired and checked by reading, not compiled, not run on hardware
//
#include <hip/hip_runtime.h>

#define BATCH 4
#define SEQ   2048
#define DIM   1024
#define MTOT  (BATCH * SEQ)   // 8192 rows

typedef __attribute__((ext_vector_type(16))) __bf16 bf16x16;
typedef __attribute__((ext_vector_type(8)))  float  f32x8;

union Frag { bf16x16 v; uint4 q[2]; };

static __device__ __forceinline__ f32x8 wmma_bf16(bf16x16 a, bf16x16 b, f32x8 c) {
  // D = A(16x32 bf16) * B(32x16 bf16) + C(16x16 f32)
  return __builtin_amdgcn_wmma_f32_16x16x32_bf16(false, a, false, b, (short)0, c,
                                                 false, false);
}

static __device__ __forceinline__ unsigned short f2bf(float f) {
  unsigned int u = __float_as_uint(f);
  u += 0x7FFFu + ((u >> 16) & 1u);           // round-to-nearest-even
  return (unsigned short)(u >> 16);
}
static __device__ __forceinline__ float bf2f(unsigned short h) {
  return __uint_as_float(((unsigned int)h) << 16);
}

// A-fragment (16x32, 16-bit): lane holds row m=lane&15; halves 0-7 = K in
// [ka, ka+8), halves 8-15 = K in [ka+16, ka+24), ka = (lane>=16)*8.
static __device__ __forceinline__ bf16x16 load_afrag(const unsigned short* p0,
                                                     const unsigned short* p1) {
  Frag f;
  f.q[0] = *(const uint4*)p0;
  f.q[1] = *(const uint4*)p1;
  return f.v;
}
// B-fragment (32x16, 16-bit): lane holds col n=lane&15; halves 0-7 = K in
// [kb, kb+8), halves 8-15 = K in [kb+16, kb+24), kb = (lane>=16)*8.
static __device__ __forceinline__ bf16x16 load_bfrag(const unsigned short* p) {
  Frag f;
  f.q[0] = *(const uint4*)p;
  f.q[1] = *(const uint4*)(p + 16);
  return f.v;
}

// ---------------------------------------------------------------- convert ---
__global__ void cvt_f32_bf16_v4(const float4* __restrict__ in,
                                ushort4* __restrict__ out, int n4) {
  int i = blockIdx.x * blockDim.x + threadIdx.x;
  if (i < n4) {
    float4 f = in[i];
    ushort4 r;
    r.x = f2bf(f.x); r.y = f2bf(f.y); r.z = f2bf(f.z); r.w = f2bf(f.w);
    out[i] = r;
  }
}

// --------------------------------------------------------- GEMM C = A*W^T ---
// A: [MTOT, DIM] bf16 row-major.  W: [DIM, DIM] bf16 row-major ([out,in]).
// Block tile 128x128, 8 waves (4 M x 2 N), wave tile 32x64 (2x4 WMMA tiles).
// Software-pipelined k-loop (next fragments issued before current WMMAs).
// MODE 0: store bf16 row-major [MTOT, DIM]
// MODE 1: store bf16 transposed as [B, DIM, SEQ]   (for V)
// MODE 2: store f32 scaled by (1 - sigmoid(beta[0])) to d_out
template <int MODE>
__global__ __launch_bounds__(256) void gemm_bf16(
    const unsigned short* __restrict__ A, const unsigned short* __restrict__ W,
    void* __restrict__ Out, const float* __restrict__ beta) {
  const int lane = threadIdx.x & 31;
  const int wave = threadIdx.x >> 5;
  const int hi   = lane >> 4;
  const int c    = lane & 15;
  const int m0   = blockIdx.y * 128 + (wave & 3) * 32;
  const int n0   = blockIdx.x * 128 + (wave >> 2) * 64;
  const int ka   = hi * 8, kb = hi * 8;

  f32x8 acc[2][4];
  for (int i = 0; i < 2; ++i)
    for (int j = 0; j < 4; ++j)
      for (int v = 0; v < 8; ++v) acc[i][j][v] = 0.f;

  const unsigned short* ar0 = A + (size_t)(m0 + c) * DIM;
  const unsigned short* ar1 = A + (size_t)(m0 + 16 + c) * DIM;
  const unsigned short* br0 = W + (size_t)(n0 + c) * DIM;
  const unsigned short* br1 = W + (size_t)(n0 + 16 + c) * DIM;
  const unsigned short* br2 = W + (size_t)(n0 + 32 + c) * DIM;
  const unsigned short* br3 = W + (size_t)(n0 + 48 + c) * DIM;

  bf16x16 a0 = load_afrag(ar0 + ka, ar0 + ka + 16);
  bf16x16 a1 = load_afrag(ar1 + ka, ar1 + ka + 16);
  bf16x16 b0 = load_bfrag(br0 + kb);
  bf16x16 b1 = load_bfrag(br1 + kb);
  bf16x16 b2 = load_bfrag(br2 + kb);
  bf16x16 b3 = load_bfrag(br3 + kb);

  for (int k0 = 0; k0 < DIM; k0 += 32) {
    const int kn = (k0 + 32 < DIM) ? (k0 + 32) : k0;   // clamped tail prefetch
    bf16x16 na0 = load_afrag(ar0 + kn + ka, ar0 + kn + ka + 16);
    bf16x16 na1 = load_afrag(ar1 + kn + ka, ar1 + kn + ka + 16);
    bf16x16 nb0 = load_bfrag(br0 + kn + kb);
    bf16x16 nb1 = load_bfrag(br1 + kn + kb);
    bf16x16 nb2 = load_bfrag(br2 + kn + kb);
    bf16x16 nb3 = load_bfrag(br3 + kn + kb);

    acc[0][0] = wmma_bf16(a0, b0, acc[0][0]);
    acc[1][0] = wmma_bf16(a1, b0, acc[1][0]);
    acc[0][1] = wmma_bf16(a0, b1, acc[0][1]);
    acc[1][1] = wmma_bf16(a1, b1, acc[1][1]);
    acc[0][2] = wmma_bf16(a0, b2, acc[0][2]);
    acc[1][2] = wmma_bf16(a1, b2, acc[1][2]);
    acc[0][3] = wmma_bf16(a0, b3, acc[0][3]);
    acc[1][3] = wmma_bf16(a1, b3, acc[1][3]);

    a0 = na0; a1 = na1;
    b0 = nb0; b1 = nb1; b2 = nb2; b3 = nb3;
  }

  if (MODE == 2) {
    const float g  = 1.f / (1.f + __expf(-beta[0]));
    const float sc = 1.f - g;
    float* out = (float*)Out;
    for (int i = 0; i < 2; ++i)
      for (int j = 0; j < 4; ++j)
        for (int v = 0; v < 8; ++v) {
          int m = m0 + 16 * i + v + 8 * hi;
          int n = n0 + 16 * j + c;
          out[(size_t)m * DIM + n] = acc[i][j][v] * sc;
        }
  } else if (MODE == 1) {
    unsigned short* out = (unsigned short*)Out;
    for (int i = 0; i < 2; ++i)
      for (int j = 0; j < 4; ++j)
        for (int v = 0; v < 8; ++v) {
          int m = m0 + 16 * i + v + 8 * hi;
          int n = n0 + 16 * j + c;
          int b = m >> 11, l = m & (SEQ - 1);
          out[((size_t)b * DIM + n) * SEQ + l] = f2bf(acc[i][j][v]);
        }
  } else {
    unsigned short* out = (unsigned short*)Out;
    for (int i = 0; i < 2; ++i)
      for (int j = 0; j < 4; ++j)
        for (int v = 0; v < 8; ++v) {
          int m = m0 + 16 * i + v + 8 * hi;
          int n = n0 + 16 * j + c;
          out[(size_t)m * DIM + n] = f2bf(acc[i][j][v]);
        }
  }
}

// --------------------------------------------------------- in-place RoPE ---
__global__ void rope_kernel(unsigned short* __restrict__ Q,
                            unsigned short* __restrict__ K,
                            const float* __restrict__ cosT,
                            const float* __restrict__ sinT) {
  int i = blockIdx.x * blockDim.x + threadIdx.x;  // over B*L*(D/2)
  if (i >= BATCH * SEQ * (DIM / 2)) return;
  int d = i & 511;
  int r = i >> 9;            // b*SEQ + l
  int l = r & (SEQ - 1);
  float c1 = cosT[l * DIM + d],       s1 = sinT[l * DIM + d];
  float c2 = cosT[l * DIM + d + 512], s2 = sinT[l * DIM + d + 512];
  size_t o1 = (size_t)r * DIM + d, o2 = o1 + 512;
  float q1 = bf2f(Q[o1]), q2 = bf2f(Q[o2]);
  Q[o1] = f2bf(q1 * c1 - q2 * s1);
  Q[o2] = f2bf(q2 * c2 + q1 * s2);
  float k1 = bf2f(K[o1]), k2 = bf2f(K[o2]);
  K[o1] = f2bf(k1 * c1 - k2 * s1);
  K[o2] = f2bf(k2 * c2 + k1 * s2);
}

// --------------------------------------------- flash causal SDPA (1 head) ---
// Block: 16 query rows, 8 waves; wave w owns head-dim slice [128w, 128w+128).
// Qb,Kb: [B,SEQ,DIM] bf16.  Vt: [B,DIM,SEQ] bf16.  Ab out: [B,SEQ,DIM] bf16.
__global__ __launch_bounds__(256) void attn_kernel(
    const unsigned short* __restrict__ Qb, const unsigned short* __restrict__ Kb,
    const unsigned short* __restrict__ Vt, unsigned short* __restrict__ Ab) {
  // partial scores laid out [row][col][wave] so the reduction is two
  // contiguous ds_load_b128 per (row,col) instead of 8 scalar reads.
  __shared__ float ldsS[16][32][8];          // 16 KB
  __shared__ unsigned short ldsP[16][32];    // softmaxed P tile (bf16)

  const int lane  = threadIdx.x & 31;
  const int wave  = threadIdx.x >> 5;
  const int hi    = lane >> 4;
  const int c     = lane & 15;
  const int q0    = blockIdx.x * 16;
  const int b     = blockIdx.y;
  const int dbase = wave * 128;
  const int ka = hi * 8, kb = hi * 8;
  const float scale = 0.03125f;              // 1/sqrt(1024)

  f32x8 acc[8];
  for (int t = 0; t < 8; ++t)
    for (int v = 0; v < 8; ++v) acc[t][v] = 0.f;
  float mrow[8], lrow[8];
  for (int v = 0; v < 8; ++v) { mrow[v] = -1e30f; lrow[v] = 0.f; }

  bf16x16 qf[4];
  const unsigned short* qrow = Qb + (size_t)(b * SEQ + q0 + c) * DIM;
  for (int i = 0; i < 4; ++i) {
    int d0 = dbase + 32 * i + ka;
    qf[i] = load_afrag(qrow + d0, qrow + d0 + 16);
  }
  const unsigned short* vb = Vt + (size_t)b * DIM * SEQ;

  for (int j0 = 0; j0 <= q0; j0 += 32) {
    // ---- batch-load all K fragments for this key tile
    int kr0 = j0 + c;       if (kr0 > SEQ - 1) kr0 = SEQ - 1;  // masked anyway
    int kr1 = j0 + 16 + c;  if (kr1 > SEQ - 1) kr1 = SEQ - 1;
    const unsigned short* krow0 = Kb + (size_t)(b * SEQ + kr0) * DIM;
    const unsigned short* krow1 = Kb + (size_t)(b * SEQ + kr1) * DIM;
    bf16x16 kf0[4], kf1[4];
    for (int i = 0; i < 4; ++i) {
      int d0 = dbase + 32 * i + kb;
      kf0[i] = load_bfrag(krow0 + d0);
      kf1[i] = load_bfrag(krow1 + d0);
    }
    // ---- issue V fragment loads early; they complete during softmax
    int vk = j0 + kb; if (vk > SEQ - 16) vk = SEQ - 16;   // fully-masked tail
    bf16x16 vf[8];
    for (int nt = 0; nt < 8; ++nt) {
      int d = dbase + 16 * nt + c;
      vf[nt] = load_bfrag(vb + (size_t)d * SEQ + vk);
    }

    // ---- partial S = Q_slice * K_slice^T  (2 key sub-tiles of 16)
    f32x8 s0, s1;
    for (int v = 0; v < 8; ++v) { s0[v] = 0.f; s1[v] = 0.f; }
    for (int i = 0; i < 4; ++i) {
      s0 = wmma_bf16(qf[i], kf0[i], s0);
      s1 = wmma_bf16(qf[i], kf1[i], s1);
    }
    for (int v = 0; v < 8; ++v) {
      int r = v + 8 * hi;
      ldsS[r][c][wave]      = s0[v];
      ldsS[r][16 + c][wave] = s1[v];
    }
    __syncthreads();

    // ---- reduce across waves (vector LDS reads), mask, online softmax
    float p0a[8], p1a[8], alph[8];
    for (int v = 0; v < 8; ++v) {
      int r = v + 8 * hi;
      const float4* p0v = (const float4*)&ldsS[r][c][0];
      const float4* p1v = (const float4*)&ldsS[r][16 + c][0];
      float4 x0 = p0v[0], x1 = p0v[1];
      float4 y0 = p1v[0], y1 = p1v[1];
      float a0 = (x0.x + x0.y) + (x0.z + x0.w) + (x1.x + x1.y) + (x1.z + x1.w);
      float a1 = (y0.x + y0.y) + (y0.z + y0.w) + (y1.x + y1.y) + (y1.z + y1.w);
      a0 *= scale; a1 *= scale;
      int qg = q0 + r;
      if (j0 + c > qg)      a0 = -1e30f;
      if (j0 + 16 + c > qg) a1 = -1e30f;
      float rmax = fmaxf(a0, a1);
      for (int m = 1; m < 16; m <<= 1) rmax = fmaxf(rmax, __shfl_xor(rmax, m, 32));
      float mn = fmaxf(mrow[v], rmax);
      float al = __expf(mrow[v] - mn);
      float p0 = __expf(a0 - mn);
      float p1 = __expf(a1 - mn);
      float rs = p0 + p1;
      for (int m = 1; m < 16; m <<= 1) rs += __shfl_xor(rs, m, 32);
      lrow[v] = lrow[v] * al + rs;
      mrow[v] = mn;
      alph[v] = al; p0a[v] = p0; p1a[v] = p1;
    }
    for (int t = 0; t < 8; ++t)
      for (int v = 0; v < 8; ++v) acc[t][v] *= alph[v];

    if (wave == 0) {
      for (int v = 0; v < 8; ++v) {
        int r = v + 8 * hi;
        ldsP[r][c]      = f2bf(p0a[v]);
        ldsP[r][16 + c] = f2bf(p1a[v]);
      }
    }
    __syncthreads();

    // ---- acc += P(16x32) * V(32 x 128-slice); V frags already in flight
    bf16x16 pf = load_afrag(&ldsP[c][ka], &ldsP[c][16 + ka]);
    for (int nt = 0; nt < 8; ++nt)
      acc[nt] = wmma_bf16(pf, vf[nt], acc[nt]);
    __syncthreads();
  }

  float inv[8];
  for (int v = 0; v < 8; ++v) inv[v] = 1.f / lrow[v];
  for (int nt = 0; nt < 8; ++nt)
    for (int v = 0; v < 8; ++v) {
      int r = v + 8 * hi;
      int d = dbase + 16 * nt + c;
      Ab[(size_t)(b * SEQ + q0 + r) * DIM + d] = f2bf(acc[nt][v] * inv[v]);
    }
}

// ------------------------------------------------------------------ launch --
extern "C" void kernel_launch(void* const* d_in, const int* in_sizes, int n_in,
                              void* d_out, int out_size, void* d_ws, size_t ws_size,
                              hipStream_t stream) {
  const float* x    = (const float*)d_in[0];
  const float* cosT = (const float*)d_in[1];
  const float* sinT = (const float*)d_in[2];
  const float* Wq   = (const float*)d_in[3];
  const float* Wk   = (const float*)d_in[4];
  const float* Wv   = (const float*)d_in[5];
  // d_in[6] = Wmp : mathematically dead (M == 0  =>  Amem == 0)
  const float* beta = (const float*)d_in[7];
  const float* Wout = (const float*)d_in[8];

  char* ws = (char*)d_ws;
  const size_t MB = 1u << 20;
  unsigned short* xb  = (unsigned short*)(ws + 0 * MB);   // 16 MB
  unsigned short* Qb  = (unsigned short*)(ws + 16 * MB);  // 16 MB
  unsigned short* Kb  = (unsigned short*)(ws + 32 * MB);  // 16 MB
  unsigned short* Vt  = (unsigned short*)(ws + 48 * MB);  // 16 MB  [B,D,L]
  unsigned short* Ab  = (unsigned short*)(ws + 64 * MB);  // 16 MB
  unsigned short* wqb = (unsigned short*)(ws + 80 * MB);  // 2 MB
  unsigned short* wkb = (unsigned short*)(ws + 82 * MB);
  unsigned short* wvb = (unsigned short*)(ws + 84 * MB);
  unsigned short* wob = (unsigned short*)(ws + 86 * MB);

  const int nX4 = MTOT * DIM / 4;  // 2.1M float4
  const int nW4 = DIM * DIM / 4;   // 256K float4
  cvt_f32_bf16_v4<<<(nX4 + 255) / 256, 256, 0, stream>>>(
      (const float4*)x, (ushort4*)xb, nX4);
  cvt_f32_bf16_v4<<<(nW4 + 255) / 256, 256, 0, stream>>>(
      (const float4*)Wq, (ushort4*)wqb, nW4);
  cvt_f32_bf16_v4<<<(nW4 + 255) / 256, 256, 0, stream>>>(
      (const float4*)Wk, (ushort4*)wkb, nW4);
  cvt_f32_bf16_v4<<<(nW4 + 255) / 256, 256, 0, stream>>>(
      (const float4*)Wv, (ushort4*)wvb, nW4);
  cvt_f32_bf16_v4<<<(nW4 + 255) / 256, 256, 0, stream>>>(
      (const float4*)Wout, (ushort4*)wob, nW4);

  dim3 gg(DIM / 128, MTOT / 128);  // (8, 64)
  gemm_bf16<0><<<gg, 256, 0, stream>>>(xb, wqb, (void*)Qb, nullptr);
  gemm_bf16<0><<<gg, 256, 0, stream>>>(xb, wkb, (void*)Kb, nullptr);
  gemm_bf16<1><<<gg, 256, 0, stream>>>(xb, wvb, (void*)Vt, nullptr);

  const int nR = BATCH * SEQ * (DIM / 2);
  rope_kernel<<<(nR + 255) / 256, 256, 0, stream>>>(Qb, Kb, cosT, sinT);

  attn_kernel<<<dim3(SEQ / 16, BATCH), 256, 0, stream>>>(Qb, Kb, Vt, Ab);

  gemm_bf16<2><<<gg, 256, 0, stream>>>(Ab, wob, d_out, beta);
}
